// CharRNN_35485019800354
// MI455X (gfx1250) — matrix operand order repaired and checked by reading, not compiled, not run on hardware
//
#include <hip/hip_runtime.h>
#include <cstddef>

// ---------------------------------------------------------------------------
// Problem dimensions (char-RNN: 2-layer LSTM + tied-softmax LM head)
// ---------------------------------------------------------------------------
constexpr int V   = 32000;   // vocab
constexpr int H   = 1024;    // rnn size
constexpr int Bsz = 32;      // batch
constexpr int T   = 256;     // seq length
constexpr int K2  = 2 * H;   // 2048 (LSTM GEMM K)
constexpr int G4  = 4 * H;   // 4096 (LSTM GEMM N)
constexpr int BT  = Bsz * T; // 8192 rows of logits

// LDS row padding: +8 bf16 (16 B) -> row stride mod 64 banks = 4 dwords,
// so the 16 M-lanes of a fragment read disjoint bank quads.
constexpr int AP = K2 + 8;   // 2056 (lstm A stage row stride, ushorts)
constexpr int HP = H + 8;    // 1032 (logits A stage row stride, ushorts)

// ---------------------------------------------------------------------------
// CDNA5 WMMA vector types
// ---------------------------------------------------------------------------
typedef __attribute__((ext_vector_type(16))) __bf16 v16bf;
typedef __attribute__((ext_vector_type(8)))  __bf16 v8bf;
typedef __attribute__((ext_vector_type(8)))  float  v8f;

union ABfrag {
    v16bf v;
    struct { v8bf lo; v8bf hi; } h;   // lo: K e=0..7 chunk, hi: K e=8..15 chunk
};

// float -> bf16 bits, round-to-nearest-even
static __device__ __forceinline__ unsigned short f2bf(float f) {
    unsigned int u = __float_as_uint(f);
    u = (u + 0x7FFFu + ((u >> 16) & 1u)) >> 16;
    return (unsigned short)u;
}
static __device__ __forceinline__ float sigm(float x) {
    return 1.0f / (1.0f + __expf(-x));
}

// ---------------------------------------------------------------------------
// gfx1250 async global->LDS copy (16 B per lane), tracked by ASYNCcnt.
// Generic LDS pointer low 32 bits == wave-relative LDS byte address.
// ---------------------------------------------------------------------------
static __device__ __forceinline__ void async_cp16(void* lds_dst, const void* gsrc) {
    unsigned loff = (unsigned)(unsigned long long)lds_dst;
    asm volatile("global_load_async_to_lds_b128 %0, %1, off"
                 :: "v"(loff), "v"(gsrc) : "memory");
}
static __device__ __forceinline__ void async_join() {
    asm volatile("s_wait_asynccnt 0x0" ::: "memory");
}

// ---------------------------------------------------------------------------
// 0) zero LSTM state (c0,c1 fp32; h0,h1 bf16) -- rerun every call: determinism
// ---------------------------------------------------------------------------
__global__ __launch_bounds__(256) void init_state(float* c0, float* c1,
                                                  unsigned short* h0,
                                                  unsigned short* h1) {
    int i = blockIdx.x * 256 + threadIdx.x;     // 32768 = Bsz*H
    c0[i] = 0.0f; c1[i] = 0.0f; h0[i] = 0; h1[i] = 0;
}

// ---------------------------------------------------------------------------
// 1) embedding gather -> x_bf [B, T, H] bf16
// ---------------------------------------------------------------------------
__global__ __launch_bounds__(256) void gather_embed(const int* __restrict__ ids,
                                                    const float* __restrict__ emb,
                                                    unsigned short* __restrict__ xbf) {
    int tid = blockIdx.x * 256 + threadIdx.x;   // Bsz*T*H threads
    int u  = tid & (H - 1);
    int bt = tid >> 10;
    xbf[tid] = f2bf(emb[(size_t)ids[bt] * H + u]);
}

// ---------------------------------------------------------------------------
// 2) pack LSTM weight [2048 x 4096] fp32 row-major into WMMA B-fragment order:
//    Wp[((ntile*64 + ktile)*32 + lane)*16 + e] = bf16(W[k, n])
//    k = ktile*32 + (lane>=16 ? 8 : 0) + (e&7) + (e>=8 ? 16 : 0),   n = ntile*16 + (lane&15)
// ---------------------------------------------------------------------------
__global__ __launch_bounds__(256) void pack_w(const float* __restrict__ W,
                                              unsigned short* __restrict__ Wp) {
    int tid   = blockIdx.x * 256 + threadIdx.x;   // 2048*4096 threads
    int e     = tid & 15;
    int lane  = (tid >> 4) & 31;
    int ktile = (tid >> 9) & 63;
    int ntile = tid >> 15;                        // 0..255
    int k = ktile * 32 + (lane >> 4) * 8 + (e & 7) + (e >> 3) * 16;
    int n = ntile * 16 + (lane & 15);
    Wp[tid] = f2bf(W[(size_t)k * G4 + n]);
}

// ---------------------------------------------------------------------------
// 3) pack embedding^T (B operand of logits GEMM, K=1024, N=32000):
//    EP[((ntile*32 + ktile)*32 + lane)*16 + e] = bf16(emb[n, k])
// ---------------------------------------------------------------------------
__global__ __launch_bounds__(256) void pack_emb(const float* __restrict__ emb,
                                                unsigned short* __restrict__ EP) {
    int tid   = blockIdx.x * 256 + threadIdx.x;   // 2000*32*32*16 = 32,768,000
    int e     = tid & 15;
    int lane  = (tid >> 4) & 31;
    int ktile = (tid >> 9) & 31;
    int ntile = tid >> 14;                        // 0..1999
    int k = ktile * 32 + (lane >> 4) * 8 + (e & 7) + (e >> 3) * 16;
    int n = ntile * 16 + (lane & 15);
    EP[tid] = f2bf(emb[(size_t)n * H + k]);
}

// ---------------------------------------------------------------------------
// 4) per-step LSTM GEMM: Z[32,4096] = [A_low | A_high] (32 x 2048) @ Wp
//    Block = 8 waves; the shared 32x2048 A operand is async-staged into LDS
//    once, then every wave computes BOTH 16-row M-tiles for its own N-tile
//    (two independent WMMA accumulators sharing one B fragment).
//    Grid: 32 blocks (8 N-tiles per block, 256 N-tiles total).
// ---------------------------------------------------------------------------
__global__ __launch_bounds__(256) void lstm_gemm(const unsigned short* __restrict__ Alow,
                                                 int sLow,
                                                 const unsigned short* __restrict__ Ahigh,
                                                 const unsigned short* __restrict__ Wp,
                                                 float* __restrict__ Z) {
    __shared__ unsigned short Ald[Bsz * AP];            // ~128.5 KB
    const int tid = threadIdx.x;

    // async-stage A: 32 rows x 2048 bf16 = 8192 chunks of 16 B
    for (int c = tid; c < Bsz * (K2 / 8); c += 256) {   // 256 chunks per row
        int m = c >> 8;
        int k = (c & 255) * 8;
        const unsigned short* src = (k < H) ? (Alow  + (size_t)m * sLow + k)
                                            : (Ahigh + (size_t)m * H + (k - H));
        async_cp16(&Ald[m * AP + k], src);
    }
    async_join();
    __syncthreads();

    const int lane  = tid & 31;
    const int w     = tid >> 5;
    const int ntile = blockIdx.x * 8 + w;               // 0..255
    const int mlo   = lane & 15;
    const int koff  = (lane >> 4) * 8;                  // ISA 16-bit A layout
    const unsigned short* wp = Wp + (size_t)ntile * 64 * 512;

    v8f acc0 = {}, acc1 = {};
    ABfrag a0, a1;
    for (int kb8 = 0; kb8 < 64; kb8 += 8) {
        __builtin_prefetch(wp + (size_t)(kb8 + 8) * 512, 0, 1);  // global_prefetch
#pragma unroll
        for (int ki = 0; ki < 8; ++ki) {
            const int ko = kb8 + ki;
            const int kb = ko * 32 + koff;
            a0.h.lo = *(const v8bf*)&Ald[(size_t)mlo * AP + kb];
            a0.h.hi = *(const v8bf*)&Ald[(size_t)mlo * AP + kb + 16];
            a1.h.lo = *(const v8bf*)&Ald[(size_t)(mlo + 16) * AP + kb];
            a1.h.hi = *(const v8bf*)&Ald[(size_t)(mlo + 16) * AP + kb + 16];
            v16bf b = *(const v16bf*)(wp + (size_t)ko * 512 + lane * 16);
            acc0 = __builtin_amdgcn_wmma_f32_16x16x32_bf16(false, a0.v, false, b,
                                                           (short)0, acc0, false, false);
            acc1 = __builtin_amdgcn_wmma_f32_16x16x32_bf16(false, a1.v, false, b,
                                                           (short)0, acc1, false, false);
        }
    }
    // ISA f32 C/D layout: lanes 0-15 -> M = r, lanes 16-31 -> M = r + 8
    const int mrow = (lane >> 4) * 8;
    const int ncol = ntile * 16 + (lane & 15);
#pragma unroll
    for (int r = 0; r < 8; ++r) {
        Z[(size_t)(mrow + r)      * G4 + ncol] = acc0[r];
        Z[(size_t)(mrow + r + 16) * G4 + ncol] = acc1[r];
    }
}

// ---------------------------------------------------------------------------
// 5) LSTM gate nonlinearity (TF LSTMCell, forget_bias = 1.0)
// ---------------------------------------------------------------------------
__global__ __launch_bounds__(256) void lstm_gate(const float* __restrict__ Z,
                                                 const float* __restrict__ bias,
                                                 float* __restrict__ c,
                                                 unsigned short* __restrict__ hbf,
                                                 unsigned short* __restrict__ hseq,
                                                 int t) {
    int idx = blockIdx.x * 256 + threadIdx.x;   // 0..32767
    int b = idx >> 10, u = idx & (H - 1);
    const float* z = Z + (size_t)b * G4;
    float gi = z[u]         + bias[u];
    float gj = z[u + H]     + bias[u + H];
    float gf = z[u + 2 * H] + bias[u + 2 * H];
    float go = z[u + 3 * H] + bias[u + 3 * H];
    float c2 = c[idx] * sigm(gf + 1.0f) + sigm(gi) * tanhf(gj);
    float h2 = sigm(go) * tanhf(c2);
    c[idx] = c2;
    unsigned short hb = f2bf(h2);
    hbf[idx] = hb;
    if (hseq) hseq[((size_t)b * T + t) * H + u] = hb;   // layer-1 output sequence
}

// ---------------------------------------------------------------------------
// 6) logits GEMM: [8192,1024] bf16 @ packed emb^T -> fp32 logits + softmax_b
//    Block = one 16-row M-tile (async-staged into LDS once) x 40 N-tiles;
//    each wave: 5 N-tiles off one LDS A fragment (5 independent accumulators).
//    Grid: 512 M-tiles x 50 N-groups = 25600 blocks.
// ---------------------------------------------------------------------------
__global__ __launch_bounds__(256) void logits_gemm(const unsigned short* __restrict__ Hseq,
                                                   const unsigned short* __restrict__ EP,
                                                   const float* __restrict__ sb,
                                                   float* __restrict__ out) {
    __shared__ unsigned short Ald[16 * HP];             // ~33 KB
    const int tid   = threadIdx.x;
    const int mtile = blockIdx.x / 50;                  // 0..511
    const int ngrp  = blockIdx.x % 50;                  // 40 N-tiles per block

    // async-stage A: 16 rows x 1024 bf16 = 2048 chunks of 16 B
    for (int c = tid; c < 16 * (H / 8); c += 256) {     // 128 chunks per row
        int m = c >> 7;
        int k = (c & 127) * 8;
        async_cp16(&Ald[m * HP + k], Hseq + (size_t)(mtile * 16 + m) * H + k);
    }
    async_join();
    __syncthreads();

    const int lane = tid & 31;
    const int w    = tid >> 5;
    const int n0   = ngrp * 40 + w * 5;                 // first of 5 N-tiles
    const int mlo  = lane & 15;
    const int koff = (lane >> 4) * 8;
    const unsigned short* epb = EP + (size_t)n0 * 32 * 512;

    v8f acc[5] = {{}, {}, {}, {}, {}};
    ABfrag a;
    for (int ko = 0; ko < 32; ++ko) {
        a.h.lo = *(const v8bf*)&Ald[(size_t)mlo * HP + ko * 32 + koff];
        a.h.hi = *(const v8bf*)&Ald[(size_t)mlo * HP + ko * 32 + koff + 16];
#pragma unroll
        for (int j = 0; j < 5; ++j) {
            v16bf b = *(const v16bf*)(epb + ((size_t)j * 32 + ko) * 512 + lane * 16);
            acc[j] = __builtin_amdgcn_wmma_f32_16x16x32_bf16(false, a.v, false, b,
                                                             (short)0, acc[j], false, false);
        }
    }
    const int mrow = mtile * 16 + (lane >> 4) * 8;
#pragma unroll
    for (int j = 0; j < 5; ++j) {
        int n = (n0 + j) * 16 + (lane & 15);
        float bn = sb[n];
#pragma unroll
        for (int r = 0; r < 8; ++r)
            out[(size_t)(mrow + r) * V + n] = acc[j][r] + bn;
    }
}

// ---------------------------------------------------------------------------
// 7) per-row log-softmax NLL (one block per row; logits already in d_out)
// ---------------------------------------------------------------------------
__global__ __launch_bounds__(256) void row_loss(const float* __restrict__ logits,
                                                const int* __restrict__ labels,
                                                float* __restrict__ losses) {
    __shared__ float red[256];
    const int row = blockIdx.x;
    const float* lr = logits + (size_t)row * V;
    float mx = -3.402823466e38f;
    for (int v = threadIdx.x; v < V; v += 256) mx = fmaxf(mx, lr[v]);
    red[threadIdx.x] = mx; __syncthreads();
    for (int s = 128; s > 0; s >>= 1) {
        if (threadIdx.x < s) red[threadIdx.x] = fmaxf(red[threadIdx.x], red[threadIdx.x + s]);
        __syncthreads();
    }
    mx = red[0]; __syncthreads();
    float sum = 0.0f;
    for (int v = threadIdx.x; v < V; v += 256) sum += __expf(lr[v] - mx);
    red[threadIdx.x] = sum; __syncthreads();
    for (int s = 128; s > 0; s >>= 1) {
        if (threadIdx.x < s) red[threadIdx.x] += red[threadIdx.x + s];
        __syncthreads();
    }
    if (threadIdx.x == 0)
        losses[row] = (mx + __logf(red[0])) - lr[labels[row]];
}

// deterministic final reduction: cost = sum(losses) / B
__global__ __launch_bounds__(256) void final_cost(const float* __restrict__ losses,
                                                  float* __restrict__ out_cost) {
    __shared__ float red[256];
    float s = 0.0f;
    for (int i = threadIdx.x; i < BT; i += 256) s += losses[i];
    red[threadIdx.x] = s; __syncthreads();
    for (int k = 128; k > 0; k >>= 1) {
        if (threadIdx.x < k) red[threadIdx.x] += red[threadIdx.x + k];
        __syncthreads();
    }
    if (threadIdx.x == 0) *out_cost = red[0] / (float)Bsz;
}

// ---------------------------------------------------------------------------
// host launcher
// ---------------------------------------------------------------------------
extern "C" void kernel_launch(void* const* d_in, const int* in_sizes, int n_in,
                              void* d_out, int out_size, void* d_ws, size_t ws_size,
                              hipStream_t stream) {
    const int*   input_data = (const int*)  d_in[0];
    const int*   targets    = (const int*)  d_in[1];
    const float* embedding  = (const float*)d_in[2];
    const float* softmax_b  = (const float*)d_in[3];
    const float* W0         = (const float*)d_in[4];
    const float* b0         = (const float*)d_in[5];
    const float* W1         = (const float*)d_in[6];
    const float* b1         = (const float*)d_in[7];

    float* logits = (float*)d_out;                    // [8192, 32000]
    float* cost   = logits + (size_t)BT * V;          // scalar

    // workspace carve-up (~134 MB, all sizes 256B-aligned by construction)
    char* ws = (char*)d_ws;
    unsigned short* x_bf = (unsigned short*)ws; ws += (size_t)BT * H * 2;       // 16 MB
    unsigned short* W0p  = (unsigned short*)ws; ws += (size_t)K2 * G4 * 2;      // 16 MB
    unsigned short* W1p  = (unsigned short*)ws; ws += (size_t)K2 * G4 * 2;      // 16 MB
    unsigned short* EP   = (unsigned short*)ws; ws += (size_t)V * H * 2;        // 64 MB
    unsigned short* hseq = (unsigned short*)ws; ws += (size_t)BT * H * 2;       // 16 MB
    unsigned short* h0   = (unsigned short*)ws; ws += (size_t)Bsz * H * 2;
    unsigned short* h1   = (unsigned short*)ws; ws += (size_t)Bsz * H * 2;
    float* c0     = (float*)ws; ws += (size_t)Bsz * H * 4;
    float* c1     = (float*)ws; ws += (size_t)Bsz * H * 4;
    float* z      = (float*)ws; ws += (size_t)Bsz * G4 * 4;
    float* losses = (float*)ws; ws += (size_t)BT * 4;

    // phase 0: deterministic state init + operand packing (weights stay in L2)
    init_state  <<<Bsz * H / 256,        256, 0, stream>>>(c0, c1, h0, h1);
    gather_embed<<<BT * H / 256,         256, 0, stream>>>(input_data, embedding, x_bf);
    pack_w      <<<K2 * G4 / 256,        256, 0, stream>>>(W0, W0p);
    pack_w      <<<K2 * G4 / 256,        256, 0, stream>>>(W1, W1p);
    pack_emb    <<<(size_t)V * H / 256,  256, 0, stream>>>(embedding, EP);

    // phase 1: sequential 2-layer LSTM, 4 kernels per timestep
    for (int t = 0; t < T; ++t) {
        lstm_gemm<<<32,  256, 0, stream>>>(x_bf + (size_t)t * H, T * H, h0, W0p, z);
        lstm_gate<<<128, 256, 0, stream>>>(z, b0, c0, h0, (unsigned short*)nullptr, 0);
        lstm_gemm<<<32,  256, 0, stream>>>(h0, H, h1, W1p, z);
        lstm_gate<<<128, 256, 0, stream>>>(z, b1, c1, h1, hseq, t);
    }

    // phase 2: tied-softmax head (store-bandwidth bound) + loss
    logits_gemm<<<25600, 256, 0, stream>>>(hseq, EP, softmax_b, logits);
    row_loss   <<<BT,    256, 0, stream>>>(logits, targets, losses);
    final_cost <<<1,     256, 0, stream>>>(losses, cost);
}
